// SelfAttentionLayer_28123445854679
// MI455X (gfx1250) — matrix operand order repaired and checked by reading, not compiled, not run on hardware
//
#include <hip/hip_runtime.h>
#include <hip/hip_bf16.h>

typedef __attribute__((ext_vector_type(16))) _Float16 v16h;
typedef __attribute__((ext_vector_type(8)))  _Float16 v8h;
typedef __attribute__((ext_vector_type(8)))  float    v8f;

#define NN   10000
#define EE   640000
#define DD   128
#define EDD  64
#define LDS_STRIDE 136   // 128 + 8 halves pad -> conflict-free ds_load_b128

// ---------- workspace layout (bytes) ----------
#define QF_OFF   0u            // 10000*128 f32
#define KF_OFF   5120000u
#define VF_OFF   10240000u
#define LG_OFF   15360000u     // 640000*8 f32 (logits, then exp() in-place)
#define M_OFF    35840000u     // 10000*8 u32 (ordered-float max)
#define S_OFF    36160000u     // 10000*8 f32 (segment sums)
#define W16_OFF  36480000u     // f16 weights: Wq|Wk|W1|W2|Wc (16384 ea) | Wre (8192)

__device__ __forceinline__ unsigned f2ord(float f) {
  unsigned u = __float_as_uint(f);
  return (u & 0x80000000u) ? ~u : (u | 0x80000000u);
}
__device__ __forceinline__ float ord2f(unsigned u) {
  return __uint_as_float((u & 0x80000000u) ? (u ^ 0x80000000u) : ~u);
}
__device__ __forceinline__ int clampN(long long v) {
  if (v < 0) v = 0;
  if (v >= NN) v = NN - 1;
  return (int)v;
}
__device__ __forceinline__ v8f wmma16(v16h a, v16h b, v8f c) {
  return __builtin_amdgcn_wmma_f32_16x16x32_f16(false, a, false, b, (short)0, c,
                                                false, false);
}
__device__ __forceinline__ float silu(float x) {
  return x / (1.0f + __expf(-x));
}

// ---------------- weight f32 -> f16 conversion ----------------
__global__ __launch_bounds__(256) void cvt_weights_kernel(
    const float* __restrict__ Wq, const float* __restrict__ Wk,
    const float* __restrict__ W1, const float* __restrict__ W2,
    const float* __restrict__ Wc, const float* __restrict__ Wre,
    _Float16* __restrict__ dst) {
  int tid = blockIdx.x * 256 + threadIdx.x;  // 352 blocks * 256 = 90112 exactly
  const float* src;
  int off;
  if (tid < 16384)       { src = Wq;  off = tid; }
  else if (tid < 32768)  { src = Wk;  off = tid - 16384; }
  else if (tid < 49152)  { src = W1;  off = tid - 32768; }
  else if (tid < 65536)  { src = W2;  off = tid - 49152; }
  else if (tid < 81920)  { src = Wc;  off = tid - 65536; }
  else                   { src = Wre; off = tid - 81920; }
  dst[tid] = (_Float16)src[off];
}

// ---------------- init max (-inf ordered) / sum buffers ----------------
__global__ __launch_bounds__(256) void init_ms_kernel(unsigned* __restrict__ m,
                                                      float* __restrict__ s) {
  int tid = blockIdx.x * 256 + threadIdx.x;
  if (tid < NN * 8) {
    m[tid] = 0x007FFFFFu;  // f2ord(-inf)
    s[tid] = 0.0f;
  }
}

// build A fragment (16x32 f16) for K-step s from an LDS tile, lane-local
__device__ __forceinline__ v16h lds_afrag(const _Float16* lds, int row, int hl,
                                          int s) {
  const _Float16* base = lds + row * LDS_STRIDE + s * 32 + hl * 16;
  v8h lo = *(const v8h*)(base);
  v8h hi = *(const v8h*)(base + 8);
  return __builtin_shufflevector(lo, hi, 0, 1, 2, 3, 4, 5, 6, 7, 8, 9, 10, 11,
                                 12, 13, 14, 15);
}

// 16x128 @ 128^T GEMM accumulate: 4 K-steps of wmma f16
__device__ __forceinline__ v8f gemm128(const v16h a[4],
                                       const _Float16* __restrict__ W16,
                                       int ng, int hl, float bias) {
  v8f acc;
#pragma unroll
  for (int i = 0; i < 8; ++i) acc[i] = bias;
#pragma unroll
  for (int s = 0; s < 4; ++s) {
    v16h b = *(const v16h*)(W16 + ng * 128 + s * 32 + hl * 16);
    acc = wmma16(a[s], b, acc);
  }
  return acc;
}

// ---------------- node projections: q, k, v = MLP(h) ----------------
__global__ __launch_bounds__(256) void node_proj_kernel(
    const float* __restrict__ h, const float* __restrict__ bq,
    const float* __restrict__ bk, const float* __restrict__ b1,
    const float* __restrict__ b2, const _Float16* __restrict__ Wq16,
    const _Float16* __restrict__ Wk16, const _Float16* __restrict__ W116,
    const _Float16* __restrict__ W216, float* __restrict__ qf,
    float* __restrict__ kf, float* __restrict__ vf) {
  __shared__ _Float16 ldsA[16 * LDS_STRIDE];
  __shared__ _Float16 ldsH[16 * LDS_STRIDE];
  const int t = threadIdx.x;
  const int node0 = blockIdx.x * 16;
  // stage 16x128 f32 h tile into LDS as f16
#pragma unroll
  for (int i = 0; i < 8; ++i) {
    int idx = i * 256 + t;
    int r = idx >> 7, c = idx & 127;
    ldsA[r * LDS_STRIDE + c] = (_Float16)h[(node0 + r) * 128 + c];
  }
  __syncthreads();

  const int lane = t & 31;
  const int wv = t >> 5;          // wave -> N-tile 0..7
  const int nlo = lane & 15;      // N within tile (also A row)
  const int hl = lane >> 4;       // half select
  const int ng = wv * 16 + nlo;   // global output feature

  v16h a[4];
#pragma unroll
  for (int s = 0; s < 4; ++s) a[s] = lds_afrag(ldsA, nlo, hl, s);

  v8f qacc = gemm128(a, Wq16, ng, hl, bq[ng]);
  v8f kacc = gemm128(a, Wk16, ng, hl, bk[ng]);
  v8f hacc = gemm128(a, W116, ng, hl, b1[ng]);

  // hidden = silu(...) -> LDS f16
#pragma unroll
  for (int r = 0; r < 8; ++r)
    ldsH[(r + hl * 8) * LDS_STRIDE + ng] = (_Float16)silu(hacc[r]);
  __syncthreads();

  v16h ah[4];
#pragma unroll
  for (int s = 0; s < 4; ++s) ah[s] = lds_afrag(ldsH, nlo, hl, s);
  v8f vacc = gemm128(ah, W216, ng, hl, b2[ng]);

#pragma unroll
  for (int r = 0; r < 8; ++r) {
    int row = node0 + r + hl * 8;
    qf[row * 128 + ng] = qacc[r];
    kf[row * 128 + ng] = kacc[r];
    vf[row * 128 + ng] = vacc[r];
  }
}

// ---------------- edge logits + segment max ----------------
__global__ __launch_bounds__(256) void edge_logits_kernel(
    const float* __restrict__ tij, const long long* __restrict__ ei,
    const float* __restrict__ qf, const float* __restrict__ kf,
    const _Float16* __restrict__ Wre16, const float* __restrict__ bre,
    float* __restrict__ logits, unsigned* __restrict__ mbuf) {
  const int t = threadIdx.x, lane = t & 31, wv = t >> 5;
  const int nlo = lane & 15, hl = lane >> 4;
  const long long e0 = (long long)(blockIdx.x * 8 + wv) * 16;

  // A fragments from t_ij (16 edges x 64), K=64 -> 2 steps
  const float* trow = tij + (e0 + nlo) * 64;
  v16h a[2];
#pragma unroll
  for (int s = 0; s < 2; ++s) {
    v16h av;
#pragma unroll
    for (int j = 0; j < 8; ++j) {
      av[j]     = (_Float16)trow[s * 32 + hl * 16 + j];
      av[j + 8] = (_Float16)trow[s * 32 + hl * 16 + 8 + j];
    }
    a[s] = av;
  }

  // destination / source node ids for my 8 edge rows (M = r + hl*8)
  int ni[8], nj[8];
#pragma unroll
  for (int r = 0; r < 8; ++r) {
    long long e = e0 + r + hl * 8;
    ni[r] = clampN(ei[EE + e]);  // n_i (row 1)
    nj[r] = clampN(ei[e]);       // n_j (row 0)
  }

  for (int head = 0; head < 8; ++head) {  // head == N-tile (d_h == 16)
    const int ng = head * 16 + nlo;
    v8f acc;
    const float bias = bre[ng];
#pragma unroll
    for (int i = 0; i < 8; ++i) acc[i] = bias;
#pragma unroll
    for (int s = 0; s < 2; ++s) {
      v16h b = *(const v16h*)(Wre16 + ng * 64 + s * 32 + hl * 16);
      acc = wmma16(a[s], b, acc);
    }
    float vals[8];
#pragma unroll
    for (int r = 0; r < 8; ++r) {
      float re = silu(acc[r]);
      float p = qf[ni[r] * 128 + ng] * kf[nj[r] * 128 + ng] * re;
      p += __shfl_xor(p, 1, 16);
      p += __shfl_xor(p, 2, 16);
      p += __shfl_xor(p, 4, 16);
      p += __shfl_xor(p, 8, 16);
      vals[r] = p;
    }
    if (nlo == 0) {  // lanes 0 and 16: write 8 edges each
#pragma unroll
      for (int r = 0; r < 8; ++r) {
        long long e = e0 + r + hl * 8;
        logits[e * 8 + head] = vals[r];
        atomicMax(mbuf + ni[r] * 8 + head, f2ord(vals[r]));
      }
    }
  }
}

// ---------------- exp(a - m) and segment sum (in-place on logits) -------
__global__ __launch_bounds__(256) void softmax_sum_kernel(
    const long long* __restrict__ ei, const unsigned* __restrict__ mbuf,
    float* __restrict__ logits, float* __restrict__ sbuf) {
  long long tid = (long long)blockIdx.x * 256 + threadIdx.x;  // < EE*8
  long long e = tid >> 3;
  int hd = (int)(tid & 7);
  int ni = clampN(ei[EE + e]);
  float m = ord2f(mbuf[ni * 8 + hd]);
  float ex = __expf(logits[tid] - m);
  logits[tid] = ex;
  atomicAdd(sbuf + ni * 8 + hd, ex);
}

// ---------------- normalize * v_j, then combine-heads GEMM ----------------
__global__ __launch_bounds__(256) void combine_kernel(
    const long long* __restrict__ ei, const float* __restrict__ exbuf,
    const float* __restrict__ sbuf, const float* __restrict__ vf,
    const _Float16* __restrict__ Wc16, const float* __restrict__ bc,
    float* __restrict__ out) {
  __shared__ _Float16 ldsS[16 * LDS_STRIDE];
  const int t = threadIdx.x;
  const long long e0 = (long long)blockIdx.x * 16;
#pragma unroll
  for (int i = 0; i < 8; ++i) {
    int idx = i * 256 + t;
    int r = idx >> 7, f = idx & 127;
    long long e = e0 + r;
    int nie = clampN(ei[EE + e]);
    int nje = clampN(ei[e]);
    int hd = f >> 4;
    float w = exbuf[e * 8 + hd] / (sbuf[nie * 8 + hd] + 1e-16f) * 0.25f;
    ldsS[r * LDS_STRIDE + f] = (_Float16)(w * vf[nje * 128 + f]);
  }
  __syncthreads();

  const int lane = t & 31, wv = t >> 5;
  const int nlo = lane & 15, hl = lane >> 4;
  const int ng = wv * 16 + nlo;

  v16h a[4];
#pragma unroll
  for (int s = 0; s < 4; ++s) a[s] = lds_afrag(ldsS, nlo, hl, s);
  v8f acc = gemm128(a, Wc16, ng, hl, bc[ng]);

#pragma unroll
  for (int r = 0; r < 8; ++r)
    out[(e0 + r + hl * 8) * 128 + ng] = acc[r];
}

extern "C" void kernel_launch(void* const* d_in, const int* in_sizes, int n_in,
                              void* d_out, int out_size, void* d_ws,
                              size_t ws_size, hipStream_t stream) {
  const float* h   = (const float*)d_in[0];
  const float* tij = (const float*)d_in[1];
  const long long* ei = (const long long*)d_in[2];
  const float* Wq = (const float*)d_in[3];  const float* bq = (const float*)d_in[4];
  const float* Wk = (const float*)d_in[5];  const float* bk = (const float*)d_in[6];
  const float* W1 = (const float*)d_in[7];  const float* b1 = (const float*)d_in[8];
  const float* W2 = (const float*)d_in[9];  const float* b2 = (const float*)d_in[10];
  const float* Wre = (const float*)d_in[11]; const float* bre = (const float*)d_in[12];
  const float* Wc = (const float*)d_in[13]; const float* bc = (const float*)d_in[14];

  char* ws = (char*)d_ws;
  float*    qf   = (float*)(ws + QF_OFF);
  float*    kf   = (float*)(ws + KF_OFF);
  float*    vf   = (float*)(ws + VF_OFF);
  float*    lg   = (float*)(ws + LG_OFF);
  unsigned* mbuf = (unsigned*)(ws + M_OFF);
  float*    sbuf = (float*)(ws + S_OFF);
  _Float16* w16  = (_Float16*)(ws + W16_OFF);
  _Float16* Wq16  = w16;
  _Float16* Wk16  = w16 + 16384;
  _Float16* W116  = w16 + 32768;
  _Float16* W216  = w16 + 49152;
  _Float16* Wc16  = w16 + 65536;
  _Float16* Wre16 = w16 + 81920;
  float* out = (float*)d_out;

  cvt_weights_kernel<<<352, 256, 0, stream>>>(Wq, Wk, W1, W2, Wc, Wre, w16);
  init_ms_kernel<<<(NN * 8 + 255) / 256, 256, 0, stream>>>(mbuf, sbuf);
  node_proj_kernel<<<NN / 16, 256, 0, stream>>>(h, bq, bk, b1, b2, Wq16, Wk16,
                                                W116, W216, qf, kf, vf);
  edge_logits_kernel<<<EE / (16 * 8), 256, 0, stream>>>(tij, ei, qf, kf, Wre16,
                                                        bre, lg, mbuf);
  softmax_sum_kernel<<<EE * 8 / 256, 256, 0, stream>>>(ei, mbuf, lg, sbuf);
  combine_kernel<<<EE / 16, 256, 0, stream>>>(ei, lg, sbuf, vf, Wc16, bc, out);
}
